// NonLocalBlock_3221225472400
// MI455X (gfx1250) — matrix operand order repaired and checked by reading, not compiled, not run on hardware
//
#include <hip/hip_runtime.h>
#include <hip/hip_bf16.h>
#include <stdint.h>

#define NPOS  6272   // T*H*W = 8*28*28
#define CCH   128    // channels
#define INTER 64
#define SPLIT 4                  // key-dimension splits (flash-decoding)
#define KSPLIT (NPOS / SPLIT)    // 1568 keys per split
#define NT (KSPLIT / 32)         // 49 key tiles per split

typedef __attribute__((ext_vector_type(16))) __bf16 v16bf;
typedef __attribute__((ext_vector_type(8)))  __bf16 v8bf;
typedef __attribute__((ext_vector_type(8)))  float  v8f;
typedef __attribute__((ext_vector_type(4)))  int    v4i;

#define GLOBAL_AS __attribute__((address_space(1)))
#define LDS_AS    __attribute__((address_space(3)))

#if defined(__has_builtin)
#if __has_builtin(__builtin_amdgcn_global_load_async_to_lds_b128)
#define ASYNC_LDS 1
#endif
#endif

__device__ __host__ inline __bf16 f2bf(float f) {
    unsigned u = __builtin_bit_cast(unsigned, f);
    unsigned r = (u + 0x7FFFu + ((u >> 16) & 1u)) >> 16;
    unsigned short s = (unsigned short)r;
    return __builtin_bit_cast(__bf16, s);
}

__device__ inline v16bf load16(const __bf16* p0, const __bf16* p1) {
    v8bf a = *(const v8bf*)p0;
    v8bf b = *(const v8bf*)p1;
    v16bf r;
#pragma unroll
    for (int i = 0; i < 8; ++i) { r[i] = a[i]; r[i + 8] = b[i]; }
    return r;
}

// 16-byte global -> LDS copy: async (ASYNCcnt) when available, sync fallback.
// AS-typed pointers built via integer casts: generic-LDS addr[31:0] == LDS
// offset (ISA 10.2 aperture rules); generic-global addr == 64-bit VA.
__device__ inline void stage16(const __bf16* gsrc, __bf16* ldst) {
#ifdef ASYNC_LDS
    GLOBAL_AS v4i* gp = (GLOBAL_AS v4i*)(unsigned long long)(uintptr_t)gsrc;
    LDS_AS    v4i* lp = (LDS_AS    v4i*)(unsigned)(uintptr_t)ldst;
    __builtin_amdgcn_global_load_async_to_lds_b128(gp, lp, 0, 0);
#else
    *(v8bf*)ldst = *(const v8bf*)gsrc;
#endif
}

// ---------------------------------------------------------------------------
// Kernel 1: Qmat[b][n][c] = bf16(x[b][c][n])   (channel-contiguous layout)
// ---------------------------------------------------------------------------
__global__ void packq_kernel(const float* __restrict__ x, __bf16* __restrict__ Qmat) {
    size_t idx = (size_t)blockIdx.x * 256 + threadIdx.x;
    int c = (int)(idx % CCH);
    size_t r = idx / CCH;
    int n = (int)(r % NPOS);
    int b = (int)(r / NPOS);
    Qmat[idx] = f2bf(x[((size_t)(b * CCH + c)) * NPOS + n]);
}

// ---------------------------------------------------------------------------
// Kernel 2: gT[b][i][n] = bf16( bg[i] + sum_c Wg[i][c] * x[b][c][n] )
// ---------------------------------------------------------------------------
__global__ void gproj_kernel(const float* __restrict__ x, const float* __restrict__ Wg,
                             const float* __restrict__ bg, __bf16* __restrict__ gT) {
    size_t idx = (size_t)blockIdx.x * 256 + threadIdx.x;
    int n = (int)(idx % NPOS);
    size_t r = idx / NPOS;
    int i = (int)(r % INTER);
    int b = (int)(r / INTER);
    float acc = bg[i];
    const float* wr = Wg + (size_t)i * CCH;
    const float* xc = x + ((size_t)b * CCH) * NPOS + n;
#pragma unroll 4
    for (int c = 0; c < CCH; ++c) acc += wr[c] * xc[(size_t)c * NPOS];
    gT[idx] = f2bf(acc);
}

// ---------------------------------------------------------------------------
// Kernel 3: flash-attention partial over one key split.
// 128 threads = 4 waves, each wave owns 16 query rows (block: 64 rows).
// K/V tiles double-buffered in LDS via async global->LDS copies.
// Writes unnormalized O plus per-row (m, l) for the merge pass.
// ---------------------------------------------------------------------------
__global__ __launch_bounds__(128) void attn_partial(
    const __bf16* __restrict__ Qmat, const __bf16* __restrict__ gT,
    float* __restrict__ Opart, float* __restrict__ Mpart, float* __restrict__ Lpart) {

    __shared__ __attribute__((aligned(32))) __bf16 Kbuf[2][32][CCH];   // 16 KB
    __shared__ __attribute__((aligned(32))) __bf16 Vbuf[2][INTER][32]; //  8 KB
    __shared__ __attribute__((aligned(32))) __bf16 pbuf[4][16][32];    //  4 KB

    const int tid  = threadIdx.x;
    const int wave = tid >> 5;
    const int lane = tid & 31;
    const int c16  = lane & 15;
    const int hi   = lane >> 4;
    const int ha   = hi * 8;

    const int ntile = NPOS / 64;                       // 98
    const int b  = blockIdx.x / (SPLIT * ntile);
    const int rs = blockIdx.x % (SPLIT * ntile);
    const int sp = rs / ntile;
    const int q0 = (rs % ntile) * 64;
    const int k0 = sp * KSPLIT;

    // --- Q fragments: A-matrix 16x32 bf16, K=128 in 4 chunks ---
    const int qrow = q0 + wave * 16 + c16;
    const __bf16* qrp = Qmat + ((size_t)(b * NPOS + qrow)) * CCH;
    v16bf qf[4];
#pragma unroll
    for (int kk = 0; kk < 4; ++kk)
        qf[kk] = load16(qrp + 32 * kk + hi * 16, qrp + 32 * kk + hi * 16 + 8);

    // cooperative stage of key tile j into buffer j&1 (6 x 16B per thread)
    auto stage = [&](int j) {
        const int kb = k0 + j * 32;
        __bf16* kd = &Kbuf[j & 1][0][0];
        __bf16* vd = &Vbuf[j & 1][0][0];
#pragma unroll
        for (int k = 0; k < 4; ++k) {                  // 32x128 bf16 K tile
            int chunk = tid + 128 * k;                 // 0..511
            int key = chunk >> 4, off = (chunk & 15) * 8;
            stage16(Qmat + ((size_t)(b * NPOS + kb + key)) * CCH + off,
                    kd + key * CCH + off);
        }
#pragma unroll
        for (int k = 0; k < 2; ++k) {                  // 64x32 bf16 V tile
            int chunk = tid + 128 * k;                 // 0..255
            int i = chunk >> 2, off = (chunk & 3) * 8;
            stage16(gT + ((size_t)(b * INTER + i)) * NPOS + kb + off,
                    vd + i * 32 + off);
        }
    };

    float m[8], l[8];
    v8f o[4];
#pragma unroll
    for (int v = 0; v < 8; ++v) { m[v] = -3.0e38f; l[v] = 0.0f; }
#pragma unroll
    for (int t = 0; t < 4; ++t) { v8f z{}; o[t] = z; }

    stage(0);
    for (int j = 0; j < NT; ++j) {
        if (j + 1 < NT) {
            stage(j + 1);                               // prefetch next tile
#ifdef ASYNC_LDS
            asm volatile("s_wait_asynccnt 0x6" ::: "memory"); // tile j landed
#endif
        } else {
#ifdef ASYNC_LDS
            asm volatile("s_wait_asynccnt 0x0" ::: "memory");
#endif
        }
        __syncthreads();                                // staged data visible to all waves
        const __bf16 (*Kt)[CCH] = Kbuf[j & 1];
        const __bf16 (*Vt)[32]  = Vbuf[j & 1];

        // --- S = Q K^T for two 16-key subtiles (8 WMMAs) ---
        v8f sAcc[2];
#pragma unroll
        for (int s2 = 0; s2 < 2; ++s2) {
            v8f acc{};
            const __bf16* krp = &Kt[16 * s2 + c16][hi * 16];
#pragma unroll
            for (int kk = 0; kk < 4; ++kk) {
                v16bf bf = *(const v16bf*)(krp + 32 * kk);
                acc = __builtin_amdgcn_wmma_f32_16x16x32_bf16(
                    false, qf[kk], false, bf, (short)0, acc, false, false);
            }
            sAcc[s2] = acc;
        }

        // --- online softmax (row reductions across 16-lane halves) ---
        float pA[8], pB[8], alpha[8];
#pragma unroll
        for (int v = 0; v < 8; ++v) {
            float mx = fmaxf(sAcc[0][v], sAcc[1][v]);
            mx = fmaxf(mx, __shfl_xor(mx, 1, 32));
            mx = fmaxf(mx, __shfl_xor(mx, 2, 32));
            mx = fmaxf(mx, __shfl_xor(mx, 4, 32));
            mx = fmaxf(mx, __shfl_xor(mx, 8, 32));
            const float mn = fmaxf(m[v], mx);
            alpha[v] = __expf(m[v] - mn);
            m[v] = mn;
            pA[v] = __expf(sAcc[0][v] - mn);
            pB[v] = __expf(sAcc[1][v] - mn);
            float rs2 = pA[v] + pB[v];
            rs2 += __shfl_xor(rs2, 1, 32);
            rs2 += __shfl_xor(rs2, 2, 32);
            rs2 += __shfl_xor(rs2, 4, 32);
            rs2 += __shfl_xor(rs2, 8, 32);
            l[v] = l[v] * alpha[v] + rs2;
        }

        // --- P: C/D layout -> A layout via per-wave LDS bounce ---
#pragma unroll
        for (int v = 0; v < 8; ++v) {
            const int row = v + 8 * hi;
            pbuf[wave][row][c16]      = f2bf(pA[v]);
            pbuf[wave][row][c16 + 16] = f2bf(pB[v]);
        }
        asm volatile("s_wait_dscnt 0" ::: "memory");    // wave lockstep store->load
        v16bf pf = load16(&pbuf[wave][c16][hi * 16], &pbuf[wave][c16][hi * 16 + 8]);

        // --- rescale O, accumulate O += P * V (4 WMMAs) ---
#pragma unroll
        for (int t = 0; t < 4; ++t)
#pragma unroll
            for (int v = 0; v < 8; ++v) o[t][v] *= alpha[v];
#pragma unroll
        for (int t = 0; t < 4; ++t) {
            v16bf vf = *(const v16bf*)(&Vt[16 * t + c16][hi * 16]);
            o[t] = __builtin_amdgcn_wmma_f32_16x16x32_bf16(
                false, pf, false, vf, (short)0, o[t], false, false);
        }
        __syncthreads();   // all waves done with buf j&1 before it is re-staged
    }

    // --- write unnormalized partials ---
#pragma unroll
    for (int t = 0; t < 4; ++t)
#pragma unroll
        for (int v = 0; v < 8; ++v) {
            const int q = q0 + wave * 16 + v + 8 * hi;
            Opart[((size_t)(b * SPLIT + sp) * NPOS + q) * INTER + 16 * t + c16] = o[t][v];
        }
    if (c16 == 0) {    // lanes 0 and 16 cover rows v and v+8
#pragma unroll
        for (int v = 0; v < 8; ++v) {
            const int q = q0 + wave * 16 + v + 8 * hi;
            const size_t idx = (size_t)(b * SPLIT + sp) * NPOS + q;
            Mpart[idx] = m[v];
            Lpart[idx] = l[v];
        }
    }
}

// ---------------------------------------------------------------------------
// Kernel 4: merge the SPLIT partials, then z = Wz*y + bz + x.
// 196 blocks x 128 threads; block covers 64 positions.
// ---------------------------------------------------------------------------
__global__ __launch_bounds__(128) void merge_kernel(
    const float* __restrict__ Opart, const float* __restrict__ Mpart,
    const float* __restrict__ Lpart, const float* __restrict__ x,
    const float* __restrict__ Wz, const float* __restrict__ bz,
    float* __restrict__ out) {

    __shared__ float ybuf[64][65];
    const int tid  = threadIdx.x;
    const int b    = blockIdx.x / (NPOS / 64);
    const int q0   = (blockIdx.x % (NPOS / 64)) * 64;
    const int qi   = tid & 63;
    const int half = tid >> 6;
    const int q    = q0 + qi;

    float ms[SPLIT], M = -3.0e38f;
#pragma unroll
    for (int s = 0; s < SPLIT; ++s) {
        ms[s] = Mpart[(size_t)(b * SPLIT + s) * NPOS + q];
        M = fmaxf(M, ms[s]);
    }
    float w[SPLIT], L = 0.0f;
#pragma unroll
    for (int s = 0; s < SPLIT; ++s) {
        w[s] = __expf(ms[s] - M);
        L += w[s] * Lpart[(size_t)(b * SPLIT + s) * NPOS + q];
    }
    const float invL = 1.0f / L;
    for (int i = half * 32; i < half * 32 + 32; ++i) {
        float acc = 0.0f;
#pragma unroll
        for (int s = 0; s < SPLIT; ++s)
            acc += w[s] * Opart[((size_t)(b * SPLIT + s) * NPOS + q) * INTER + i];
        ybuf[qi][i] = acc * invL;
    }
    __syncthreads();

    for (int c = half; c < CCH; c += 2) {
        float acc = bz[c];
        const float* wr = Wz + (size_t)c * INTER;
#pragma unroll 8
        for (int i = 0; i < INTER; ++i) acc += wr[i] * ybuf[qi][i];
        const size_t oidx = ((size_t)(b * CCH + c)) * NPOS + q;
        out[oidx] = acc + x[oidx];
    }
}

// ---------------------------------------------------------------------------
extern "C" void kernel_launch(void* const* d_in, const int* in_sizes, int n_in,
                              void* d_out, int out_size, void* d_ws, size_t ws_size,
                              hipStream_t stream) {
    const float* x  = (const float*)d_in[0];
    const float* Wg = (const float*)d_in[1];
    const float* bg = (const float*)d_in[2];
    const float* Wz = (const float*)d_in[3];
    const float* bz = (const float*)d_in[4];
    float* out = (float*)d_out;

    char* ws = (char*)d_ws;
    const size_t qbytes = (size_t)2 * NPOS * CCH * sizeof(__bf16);           // 3.2 MB
    const size_t gbytes = (size_t)2 * INTER * NPOS * sizeof(__bf16);         // 1.6 MB
    const size_t obytes = (size_t)2 * SPLIT * NPOS * INTER * sizeof(float);  // 12.8 MB
    const size_t mbytes = (size_t)2 * SPLIT * NPOS * sizeof(float);          // 0.2 MB

    __bf16* Qmat = (__bf16*)ws;
    __bf16* gT   = (__bf16*)(ws + qbytes);
    float*  Opart = (float*)(ws + qbytes + gbytes);
    float*  Mpart = (float*)(ws + qbytes + gbytes + obytes);
    float*  Lpart = (float*)(ws + qbytes + gbytes + obytes + mbytes);

    packq_kernel<<<(2 * NPOS * CCH) / 256, 256, 0, stream>>>(x, Qmat);
    gproj_kernel<<<(2 * INTER * NPOS) / 256, 256, 0, stream>>>(x, Wg, bg, gT);
    attn_partial<<<2 * SPLIT * (NPOS / 64), 128, 0, stream>>>(Qmat, gT, Opart, Mpart, Lpart);
    merge_kernel<<<2 * (NPOS / 64), 128, 0, stream>>>(Opart, Mpart, Lpart, x, Wz, bz, out);
}
